// ChargePredict_61924838474266
// MI455X (gfx1250) — matrix operand order repaired and checked by reading, not compiled, not run on hardware
//
#include <hip/hip_runtime.h>
#include <hip/hip_bf16.h>
#include <cstddef>

typedef __attribute__((ext_vector_type(16))) _Float16 v16h;
typedef __attribute__((ext_vector_type(8)))  float    v8f;

#define N_ATOMS 131072
#define HIDDEN  256
#define NMOL    1024
#define LN_EPS  1e-5f
#define QEQ_EPS 1e-6f

union Frag { unsigned int u[8]; uint4 q[2]; v16h h; };

__device__ __forceinline__ unsigned int pack2(float a, float b) {
    union { _Float16 h[2]; unsigned int u; } p;
    p.h[0] = (_Float16)a; p.h[1] = (_Float16)b;
    return p.u;
}

// halfword index of element (row m, K k) inside packed A-fragment storage
// laid out as [kt][lane(32)][v(8) u32].  A layout (16-bit, 16x32 tile):
//  lanes 0-15: V0-3 = K0-7,  V4-7 = K16-23 ; lanes 16-31: V0-3 = K8-15, V4-7 = K24-31
__device__ __forceinline__ int afrag_idx(int m, int k) {
    int kt = k >> 5, r = k & 31, g = r >> 3;
    int lane = ((g & 1) << 4) | m;
    int v = ((g >> 1) << 2) | ((r & 7) >> 1);
    return ((((kt << 5) | lane) << 3 | v) << 1) | (k & 1);
}

// ---------------- init: F_u = eps, Q_u = 0 -----------------------------------
__global__ void qeq_init(float* Qu, float* Fu) {
    int tid = blockIdx.x * blockDim.x + threadIdx.x;
    if (tid < NMOL * 16) { Qu[tid] = 0.0f; Fu[tid] = QEQ_EPS; }
}

// ---------------- pack W1 (768x256) and W2 (256x32) into f16 B-fragment layout
// B layout (16-bit, 32x16 tile): lanes 0-15 hold K=0-7,16-23, lanes 16-31 K=8-15,24-31,
// VGPR v holds K = (v>>2)*16 + kh*8 + 2(v&3), +1 (low half = even K).
// Packed with v innermost: Wp[(((kt*2+kh)*NCOL + n) * 8) + v]
__global__ void pack_weights(const float* __restrict__ W1, const float* __restrict__ W2,
                             unsigned int* __restrict__ W1p, unsigned int* __restrict__ W2p) {
    const int total1 = 24 * 2 * 256 * 8;  // 98304
    const int total2 = 8 * 2 * 32 * 8;    // 4096
    int idx = blockIdx.x * blockDim.x + threadIdx.x;
    if (idx < total1) {
        int v  = idx & 7;
        int n  = (idx >> 3) & 255;
        int kh = (idx >> 11) & 1;
        int kt = idx >> 12;
        int k0 = kt * 32 + (v >> 2) * 16 + kh * 8 + 2 * (v & 3);
        W1p[idx] = pack2(W1[k0 * 256 + n], W1[(k0 + 1) * 256 + n]);
    } else if (idx < total1 + total2) {
        int i2 = idx - total1;
        int v  = i2 & 7;
        int n  = (i2 >> 3) & 31;
        int kh = (i2 >> 8) & 1;
        int kt = i2 >> 9;
        int k0 = kt * 32 + (v >> 2) * 16 + kh * 8 + 2 * (v & 3);
        W2p[i2] = pack2(W2[k0 * 32 + n], W2[(k0 + 1) * 32 + n]);
    }
}

// ---------------- fused features + LN + MLP + segment-sum atomics -------------
__global__ __launch_bounds__(256)
void fused_mlp(const float* __restrict__ X,
               const float* __restrict__ lnw, const float* __restrict__ lnb,
               const float* __restrict__ b1v, const float* __restrict__ b2v,
               const int*   __restrict__ batch,
               const unsigned int* __restrict__ W1p,
               const unsigned int* __restrict__ W2p,
               float* __restrict__ chw, float* __restrict__ fuw,
               float* __restrict__ Qu,  float* __restrict__ Fu) {
    // LN output & h1 stored directly in WMMA A-fragment order:
    __shared__ __align__(16) unsigned int lnF[24 * 32 * 8];  // 24 KB
    __shared__ __align__(16) unsigned int h1F[8 * 32 * 8];   // 8 KB
    __shared__ float redS[16][8], redQ[16][8];
    __shared__ float muArr[16], rsArr[16];

    const int t    = threadIdx.x;
    const int wv   = t >> 5;        // wave 0..7 (wave32)
    const int lane = t & 31;
    const int nl   = lane & 15;     // row/col within 16
    const int kh   = lane >> 4;     // lane-half
    const int atom_base = blockIdx.x * 16;

    // ---- phase 1: features + LN stats (thread t owns hidden channel t) ------
    const float wI = lnw[t],       bI = lnb[t];
    const float wA = lnw[256 + t], bA = lnb[256 + t];
    const float wS = lnw[512 + t], bS = lnb[512 + t];

    float fI[16], fA[16], fS[16];
    const size_t base = ((size_t)atom_base * 256 + t) * 9;
#pragma unroll
    for (int i = 0; i < 16; ++i) {
        const float* xp = X + base + (size_t)i * (256 * 9);
        float x0 = __builtin_nontemporal_load(xp + 0);
        float x1 = __builtin_nontemporal_load(xp + 1);
        float x2 = __builtin_nontemporal_load(xp + 2);
        float x3 = __builtin_nontemporal_load(xp + 3);
        float x4 = __builtin_nontemporal_load(xp + 4);
        float x5 = __builtin_nontemporal_load(xp + 5);
        float x6 = __builtin_nontemporal_load(xp + 6);
        float x7 = __builtin_nontemporal_load(xp + 7);
        float x8 = __builtin_nontemporal_load(xp + 8);
        float I  = (x0 + x4 + x8) * (1.0f / 3.0f);
        float a01 = 0.5f * (x1 - x3), a02 = 0.5f * (x2 - x6), a12 = 0.5f * (x5 - x7);
        float nA = 2.0f * (a01 * a01 + a02 * a02 + a12 * a12);
        float s01 = 0.5f * (x1 + x3), s02 = 0.5f * (x2 + x6), s12 = 0.5f * (x5 + x7);
        float d0 = x0 - I, d1 = x4 - I, d2 = x8 - I;
        float nS = d0 * d0 + d1 * d1 + d2 * d2 + 2.0f * (s01 * s01 + s02 * s02 + s12 * s12);
        fI[i] = I; fA[i] = nA; fS[i] = nS;
        float s = I + nA + nS;
        float q = I * I + nA * nA + nS * nS;
#pragma unroll
        for (int off = 16; off >= 1; off >>= 1) {
            s += __shfl_xor(s, off, 32);
            q += __shfl_xor(q, off, 32);
        }
        if (lane == 0) { redS[i][wv] = s; redQ[i][wv] = q; }
    }
    __syncthreads();
    if (t < 16) {
        float s = 0.0f, q = 0.0f;
#pragma unroll
        for (int w2 = 0; w2 < 8; ++w2) { s += redS[t][w2]; q += redQ[t][w2]; }
        float mu  = s * (1.0f / 768.0f);
        float var = q * (1.0f / 768.0f) - mu * mu;
        muArr[t] = mu;
        rsArr[t] = rsqrtf(var + LN_EPS);
    }
    __syncthreads();
    {
        _Float16* lnH = (_Float16*)lnF;
#pragma unroll
        for (int i = 0; i < 16; ++i) {
            float mu = muArr[i], rs = rsArr[i];
            lnH[afrag_idx(i, t)]       = (_Float16)((fI[i] - mu) * rs * wI + bI);
            lnH[afrag_idx(i, 256 + t)] = (_Float16)((fA[i] - mu) * rs * wA + bA);
            lnH[afrag_idx(i, 512 + t)] = (_Float16)((fS[i] - mu) * rs * wS + bS);
        }
    }
    __syncthreads();

    // ---- GEMM1: (16x768) @ (768x256), each wave does 2 N-tiles --------------
    const int nt0 = 2 * wv, nt1 = 2 * wv + 1;
    v8f c0 = {}, c1 = {};
    for (int kt = 0; kt < 24; ++kt) {
        Frag a;
        const uint4* ap = (const uint4*)&lnF[((kt << 5) | lane) << 3];
        a.q[0] = ap[0]; a.q[1] = ap[1];
        Frag b0, b1;
        const uint4* bp0 = (const uint4*)&W1p[(((kt * 2 + kh) * 256) + nt0 * 16 + nl) << 3];
        const uint4* bp1 = (const uint4*)&W1p[(((kt * 2 + kh) * 256) + nt1 * 16 + nl) << 3];
        b0.q[0] = bp0[0]; b0.q[1] = bp0[1];
        b1.q[0] = bp1[0]; b1.q[1] = bp1[1];
        c0 = __builtin_amdgcn_wmma_f32_16x16x32_f16(false, a.h, false, b0.h,
                                                    (short)0, c0, false, false);
        c1 = __builtin_amdgcn_wmma_f32_16x16x32_f16(false, a.h, false, b1.h,
                                                    (short)0, c1, false, false);
    }
    // bias + SiLU, store h1 into A-fragment layout (C layout: VGPR r -> M=r+kh*8, N=nl+nt*16)
    {
        _Float16* h1H = (_Float16*)h1F;
        float bias0 = b1v[nt0 * 16 + nl];
        float bias1 = b1v[nt1 * 16 + nl];
#pragma unroll
        for (int r = 0; r < 8; ++r) {
            int M = r + kh * 8;
            float v0 = c0[r] + bias0;
            float v1 = c1[r] + bias1;
            float s0 = v0 * __builtin_amdgcn_rcpf(1.0f + __expf(-v0));
            float s1 = v1 * __builtin_amdgcn_rcpf(1.0f + __expf(-v1));
            h1H[afrag_idx(M, nt0 * 16 + nl)] = (_Float16)s0;
            h1H[afrag_idx(M, nt1 * 16 + nl)] = (_Float16)s1;
        }
    }
    __syncthreads();

    // ---- GEMM2: (16x256) @ (256x32) on waves 0 (charges) and 1 (f) ----------
    if (wv < 2) {
        const int nt = wv;
        v8f c = {};
        for (int kt = 0; kt < 8; ++kt) {
            Frag a;
            const uint4* ap = (const uint4*)&h1F[((kt << 5) | lane) << 3];
            a.q[0] = ap[0]; a.q[1] = ap[1];
            Frag b;
            const uint4* bp = (const uint4*)&W2p[(((kt * 2 + kh) * 32) + nt * 16 + nl) << 3];
            b.q[0] = bp[0]; b.q[1] = bp[1];
            c = __builtin_amdgcn_wmma_f32_16x16x32_f16(false, a.h, false, b.h,
                                                       (short)0, c, false, false);
        }
        float bias = b2v[nt * 16 + nl];
#pragma unroll
        for (int r = 0; r < 8; ++r) {
            int M = r + kh * 8;
            int atom = atom_base + M;
            int mol = batch[atom];
            if (nt == 0) {
                float val = c[r] + bias;
                chw[atom * 16 + nl] = val;
                atomicAdd(&Qu[mol * 16 + nl], val);
            } else {
                float fv = c[r] + bias;
                float fu = fv * fv;
                fuw[atom * 16 + nl] = fu;
                atomicAdd(&Fu[mol * 16 + nl], fu);
            }
        }
    }
}

// ---------------- finalize: QEq charge correction -----------------------------
__global__ void qeq_finalize(const float* __restrict__ chw, const float* __restrict__ fuw,
                             const float* __restrict__ Q,
                             const float* __restrict__ Qu, const float* __restrict__ Fu,
                             const int* __restrict__ batch, float* __restrict__ out) {
    int tid = blockIdx.x * blockDim.x + threadIdx.x;
    if (tid >= N_ATOMS * 16) return;
    int n = tid >> 4, j = tid & 15;
    int b = batch[n];
    float dQ = Q[n] - Qu[b * 16 + j];
    out[tid] = chw[tid] + fuw[tid] / Fu[b * 16 + j] * dQ;
}

extern "C" void kernel_launch(void* const* d_in, const int* in_sizes, int n_in,
                              void* d_out, int out_size, void* d_ws, size_t ws_size,
                              hipStream_t stream) {
    const float* X   = (const float*)d_in[0];
    const float* Q   = (const float*)d_in[1];
    const float* lnw = (const float*)d_in[2];
    const float* lnb = (const float*)d_in[3];
    const float* W1  = (const float*)d_in[4];
    const float* b1  = (const float*)d_in[5];
    const float* W2  = (const float*)d_in[6];
    const float* b2  = (const float*)d_in[7];
    const int*   bat = (const int*)d_in[8];
    float* out = (float*)d_out;

    char* ws = (char*)d_ws;
    size_t off = 0;
    unsigned int* W1p = (unsigned int*)(ws + off); off += (size_t)98304 * 4;
    unsigned int* W2p = (unsigned int*)(ws + off); off += (size_t)4096 * 4;
    float* chw = (float*)(ws + off); off += (size_t)N_ATOMS * 16 * 4;
    float* fuw = (float*)(ws + off); off += (size_t)N_ATOMS * 16 * 4;
    float* Qu  = (float*)(ws + off); off += (size_t)NMOL * 16 * 4;
    float* Fu  = (float*)(ws + off); off += (size_t)NMOL * 16 * 4;

    qeq_init<<<64, 256, 0, stream>>>(Qu, Fu);
    pack_weights<<<(98304 + 4096 + 255) / 256, 256, 0, stream>>>(W1, W2, W1p, W2p);
    fused_mlp<<<N_ATOMS / 16, 256, 0, stream>>>(X, lnw, lnb, b1, b2, bat,
                                                W1p, W2p, chw, fuw, Qu, Fu);
    qeq_finalize<<<(N_ATOMS * 16) / 256, 256, 0, stream>>>(chw, fuw, Q, Qu, Fu, bat, out);
}
